// MPNNSequenceDecoder_36490042147381
// MI455X (gfx1250) — matrix operand order repaired and checked by reading, not compiled, not run on hardware
//
#include <hip/hip_runtime.h>

#define BB 4
#define NN 1024
#define KK 48
#define DD 128
#define LL 3
#define VV 22
#define EPSF 1e-5f

typedef __attribute__((ext_vector_type(16))) _Float16 v16h;
typedef __attribute__((ext_vector_type(8)))  float    v8f;
typedef __attribute__((ext_vector_type(8)))  _Float16 h8;

#define EW_STRIDE 136   // 128 halves + 8 pad (bank-conflict avoidance)
#define XS_STRIDE 136
#define H1_STRIDE 520   // 512 + 8 pad
#define T_STRIDE  132
#define NODE_PW   (16*XS_STRIDE*2 + 16*H1_STRIDE*2)   // per-wave LDS bytes in node kernel
#define EDGE_PW   (2 * 48 * EW_STRIDE * 2)            // per-wave LDS bytes in edge kernel (abuf+mbuf)

__device__ __forceinline__ float gelu_exact(float x) {
    return 0.5f * x * (1.0f + erff(x * 0.70710678118654752440f));
}

// B-fragment (32x16 f16) from row-major [K][N]: lane = N col, halves = K rows kb+8*hi+0..7, kb+16+8*hi+0..7 (p at row kb+8*hi)
__device__ __forceinline__ v16h ld16(const _Float16* p, int stride) {
    v16h v;
    #pragma unroll
    for (int s = 0; s < 8; ++s) { v[s] = p[s * stride]; v[8 + s] = p[(16 + s) * stride]; }
    return v;
}

// A-fragment (16x32 f16) from row-major LDS tile [rows][stride] halves.
// Lane layout (ISA 7.12.2): lanes 0-15 row M, VGPR0-3 K=0..7, VGPR4-7 K=16..23;
// lanes 16-31: K=8..15 / 24..31  => two 16B chunks at kb+8*hi and kb+16+8*hi.
__device__ __forceinline__ v16h frag_a(const _Float16* buf, int stride, int row, int kb, int hi) {
    const _Float16* p = buf + row * stride + kb + 8 * hi;
    union { uint4 u[2]; v16h v; } U;
    U.u[0] = *(const uint4*)(p);
    U.u[1] = *(const uint4*)(p + 16);
    return U.v;
}

__device__ __forceinline__ v8f wmma16(v16h a, v16h b, v8f c) {
    return __builtin_amdgcn_wmma_f32_16x16x32_f16(false, a, false, b, (short)0, c, false, false);
}

// ---------------- prologue kernels ----------------

__global__ void k_rank(const int* __restrict__ dec, int* __restrict__ rank) {
    int i = blockIdx.x * blockDim.x + threadIdx.x;
    if (i < BB * NN) {
        int b = i >> 10, step = i & (NN - 1);
        rank[b * NN + dec[i]] = step;
    }
}

__global__ void k_prep(const int* __restrict__ prev, const float* __restrict__ embed,
                       const float* __restrict__ hVin,
                       float* __restrict__ hS, float* __restrict__ hV, float* __restrict__ hV0) {
    int i = blockIdx.x * blockDim.x + threadIdx.x;
    if (i < BB * NN * DD) {
        int node = i >> 7, c = i & (DD - 1);
        hS[i] = embed[(size_t)prev[node] * DD + c];
        float v = hVin[i];
        hV[i] = v; hV0[i] = v;
    }
}

__global__ void k_cvtw(const float* __restrict__ Win, const float* __restrict__ Wout,
                       _Float16* __restrict__ WinH, _Float16* __restrict__ WoutH) {
    int i = blockIdx.x * blockDim.x + threadIdx.x;
    if (i < LL * DD * 512) {
        WinH[i]  = (_Float16)Win[i];
        WoutH[i] = (_Float16)Wout[i];
    }
}

// ---------------- edge message kernel (per layer) ----------------
// grid: 512 blocks x 128 threads (4 waves). 1 wave = 1 node (loops over nodes).
// All 48 edges (3 M-tiles) processed together so every weight B-fragment
// feeds 3 WMMAs -> 3x less LDS B-fragment traffic (keeps XDL fed).

__global__ void __launch_bounds__(128) k_edge(
    const float* __restrict__ hE, const int* __restrict__ Eidx,
    const float* __restrict__ mask,
    const float* __restrict__ W1, const float* __restrict__ b1,
    const float* __restrict__ W2, const float* __restrict__ b2,
    const float* __restrict__ W3, const float* __restrict__ b3,
    const float* __restrict__ hS, const float* __restrict__ hV,
    const float* __restrict__ hV0, const int* __restrict__ rank,
    float* __restrict__ dh)
{
    extern __shared__ char smem[];
    _Float16* w1 = (_Float16*)smem;                 // 512 x EW_STRIDE
    _Float16* w2 = w1 + 512 * EW_STRIDE;            // 128 x EW_STRIDE
    _Float16* w3 = w2 + 128 * EW_STRIDE;            // 128 x EW_STRIDE
    float* bb1 = (float*)(w3 + 128 * EW_STRIDE);
    float* bb2 = bb1 + DD;
    float* bb3 = bb2 + DD;
    _Float16* pw = (_Float16*)(bb3 + DD);           // per-wave staging

    const int tid = threadIdx.x;

    // cooperative weight staging: f32 -> f16 into padded rows
    for (int idx = tid; idx < 768 * 16; idx += blockDim.x) {
        int row = idx >> 4, cb = (idx & 15) * 8;
        const float* s; _Float16* d;
        if (row < 512)      { s = W1 + (size_t)row * DD + cb;         d = w1 + row * EW_STRIDE + cb; }
        else if (row < 640) { s = W2 + (size_t)(row - 512) * DD + cb; d = w2 + (row - 512) * EW_STRIDE + cb; }
        else                { s = W3 + (size_t)(row - 640) * DD + cb; d = w3 + (row - 640) * EW_STRIDE + cb; }
        float4 a = *(const float4*)s;
        float4 b = *(const float4*)(s + 4);
        h8 h;
        h[0] = (_Float16)a.x; h[1] = (_Float16)a.y; h[2] = (_Float16)a.z; h[3] = (_Float16)a.w;
        h[4] = (_Float16)b.x; h[5] = (_Float16)b.y; h[6] = (_Float16)b.z; h[7] = (_Float16)b.w;
        *(h8*)d = h;
    }
    for (int c = tid; c < DD; c += blockDim.x) { bb1[c] = b1[c]; bb2[c] = b2[c]; bb3[c] = b3[c]; }
    __syncthreads();

    const int lane = tid & 31, wv = tid >> 5;
    const int l16 = lane & 15, hi = lane >> 4;
    _Float16* abuf = pw + (size_t)wv * (EDGE_PW / 2); // halves
    _Float16* mbuf = abuf + 48 * EW_STRIDE;

    // staging assignment: lane owns rows r0, r0+16, r0+32 at channel half ch0
    const int r0  = lane >> 1;
    const int ch0 = (lane & 1) * 64;

    const int nw = (blockDim.x >> 5) * gridDim.x;
    for (int node = blockIdx.x * (blockDim.x >> 5) + wv; node < BB * NN; node += nw) {
        const int b = node >> 10;
        const float mk = mask[node];
        const int rk = rank[node];

        // prefetch the next node's h_E rows (gfx1250 global_prefetch path)
        {
            int nxt = node + nw;
            if (nxt < BB * NN) {
                __builtin_prefetch(hE + ((size_t)nxt * KK + r0) * DD + ch0, 0, 1);
                __builtin_prefetch(hE + ((size_t)nxt * KK + r0 + 16) * DD + ch0, 0, 1);
                __builtin_prefetch(hE + ((size_t)nxt * KK + r0 + 32) * DD + ch0, 0, 1);
            }
        }

        int jj[3], at[3];
        #pragma unroll
        for (int t = 0; t < 3; ++t) {
            jj[t] = Eidx[(size_t)node * KK + (r0 + 16 * t)];
            at[t] = (rank[b * NN + jj[t]] < rk) ? 1 : 0;
        }

        float acc[8] = {0, 0, 0, 0, 0, 0, 0, 0};
        v8f C[3][8] = {};

        #pragma unroll
        for (int bl = 0; bl < 4; ++bl) {
            // stage all 48 rows of this 128-channel block
            #pragma unroll
            for (int t = 0; t < 3; ++t) {
                const int rr = r0 + 16 * t;
                const int j = jj[t], att = at[t];
                const float* src; float f;
                if (bl == 0)      { src = hV + (size_t)node * DD + ch0;                    f = 1.0f; }
                else if (bl == 1) { src = hE + ((size_t)node * KK + rr) * DD + ch0;        f = mk; }
                else if (bl == 2) { src = hS + ((size_t)b * NN + j) * DD + ch0;            f = mk * (float)att; }
                else              { src = (att ? hV : hV0) + ((size_t)b * NN + j) * DD + ch0; f = mk; }
                _Float16* drow = abuf + rr * EW_STRIDE + ch0;
                #pragma unroll
                for (int i = 0; i < 8; ++i) {
                    float4 a  = *(const float4*)(src + i * 8);
                    float4 c2 = *(const float4*)(src + i * 8 + 4);
                    h8 h;
                    h[0] = (_Float16)(a.x * f);  h[1] = (_Float16)(a.y * f);
                    h[2] = (_Float16)(a.z * f);  h[3] = (_Float16)(a.w * f);
                    h[4] = (_Float16)(c2.x * f); h[5] = (_Float16)(c2.y * f);
                    h[6] = (_Float16)(c2.z * f); h[7] = (_Float16)(c2.w * f);
                    *(h8*)(drow + i * 8) = h;
                }
            }
            // consume: 4 k-steps of 32; each B-fragment feeds 3 M-tiles
            #pragma unroll
            for (int ks = 0; ks < 4; ++ks) {
                v16h A0 = frag_a(abuf, EW_STRIDE, l16,      ks * 32, hi);
                v16h A1 = frag_a(abuf, EW_STRIDE, l16 + 16, ks * 32, hi);
                v16h A2 = frag_a(abuf, EW_STRIDE, l16 + 32, ks * 32, hi);
                const _Float16* wb = w1 + (bl * 128 + ks * 32 + 8 * hi) * EW_STRIDE + l16;
                #pragma unroll
                for (int nt = 0; nt < 8; ++nt) {
                    v16h Bf = ld16(wb + nt * 16, EW_STRIDE);
                    C[0][nt] = wmma16(A0, Bf, C[0][nt]);
                    C[1][nt] = wmma16(A1, Bf, C[1][nt]);
                    C[2][nt] = wmma16(A2, Bf, C[2][nt]);
                }
            }
        }
        // m1 = gelu(C + b1) -> mbuf (C frag: lane col = nt*16+l16, rows v+8*hi)
        #pragma unroll
        for (int mt = 0; mt < 3; ++mt)
            #pragma unroll
            for (int nt = 0; nt < 8; ++nt) {
                const int c = nt * 16 + l16;
                const float bias = bb1[c];
                #pragma unroll
                for (int v = 0; v < 8; ++v)
                    mbuf[(mt * 16 + v + 8 * hi) * EW_STRIDE + c] = (_Float16)gelu_exact(C[mt][nt][v] + bias);
            }
        // m2 = gelu(m1 @ W2 + b2)
        v8f C2[3][8] = {};
        #pragma unroll
        for (int ks = 0; ks < 4; ++ks) {
            v16h A0 = frag_a(mbuf, EW_STRIDE, l16,      ks * 32, hi);
            v16h A1 = frag_a(mbuf, EW_STRIDE, l16 + 16, ks * 32, hi);
            v16h A2 = frag_a(mbuf, EW_STRIDE, l16 + 32, ks * 32, hi);
            const _Float16* wb = w2 + (ks * 32 + 8 * hi) * EW_STRIDE + l16;
            #pragma unroll
            for (int nt = 0; nt < 8; ++nt) {
                v16h Bf = ld16(wb + nt * 16, EW_STRIDE);
                C2[0][nt] = wmma16(A0, Bf, C2[0][nt]);
                C2[1][nt] = wmma16(A1, Bf, C2[1][nt]);
                C2[2][nt] = wmma16(A2, Bf, C2[2][nt]);
            }
        }
        #pragma unroll
        for (int mt = 0; mt < 3; ++mt)
            #pragma unroll
            for (int nt = 0; nt < 8; ++nt) {
                const int c = nt * 16 + l16;
                const float bias = bb2[c];
                #pragma unroll
                for (int v = 0; v < 8; ++v)
                    mbuf[(mt * 16 + v + 8 * hi) * EW_STRIDE + c] = (_Float16)gelu_exact(C2[mt][nt][v] + bias);
            }
        // m3 = m2 @ W3  (bias folded in at the end as 48*b3)
        v8f C3[3][8] = {};
        #pragma unroll
        for (int ks = 0; ks < 4; ++ks) {
            v16h A0 = frag_a(mbuf, EW_STRIDE, l16,      ks * 32, hi);
            v16h A1 = frag_a(mbuf, EW_STRIDE, l16 + 16, ks * 32, hi);
            v16h A2 = frag_a(mbuf, EW_STRIDE, l16 + 32, ks * 32, hi);
            const _Float16* wb = w3 + (ks * 32 + 8 * hi) * EW_STRIDE + l16;
            #pragma unroll
            for (int nt = 0; nt < 8; ++nt) {
                v16h Bf = ld16(wb + nt * 16, EW_STRIDE);
                C3[0][nt] = wmma16(A0, Bf, C3[0][nt]);
                C3[1][nt] = wmma16(A1, Bf, C3[1][nt]);
                C3[2][nt] = wmma16(A2, Bf, C3[2][nt]);
            }
        }
        #pragma unroll
        for (int nt = 0; nt < 8; ++nt) {
            float s = 0.0f;
            #pragma unroll
            for (int mt = 0; mt < 3; ++mt)
                #pragma unroll
                for (int v = 0; v < 8; ++v) s += C3[mt][nt][v];
            acc[nt] = s;
        }
        // cross-half row-sum, add bias, scale, write dh
        #pragma unroll
        for (int nt = 0; nt < 8; ++nt) acc[nt] += __shfl_xor(acc[nt], 16);
        if (lane < 16) {
            #pragma unroll
            for (int nt = 0; nt < 8; ++nt) {
                const int c = nt * 16 + lane;
                dh[(size_t)node * DD + c] = (acc[nt] + 48.0f * bb3[c]) * (1.0f / 30.0f);
            }
        }
    }
}

// ---------------- node update kernel (per layer) ----------------
// 1 wave = 16 nodes (M=16 WMMA tile). LN via shfl_xor(16) reductions.

__global__ void __launch_bounds__(128) k_node(
    float* __restrict__ hVio, const float* __restrict__ dh,
    const float* __restrict__ mask,
    const _Float16* __restrict__ WinH, const float* __restrict__ b_in,
    const _Float16* __restrict__ WoutH, const float* __restrict__ b_out,
    const float* __restrict__ ln1s, const float* __restrict__ ln1b,
    const float* __restrict__ ln2s, const float* __restrict__ ln2b)
{
    extern __shared__ char smem[];
    const int tid = threadIdx.x, lane = tid & 31, wv = tid >> 5;
    const int r = lane & 15, hi = lane >> 4;
    _Float16* Xs = (_Float16*)(smem + (size_t)wv * NODE_PW);
    _Float16* H1 = Xs + 16 * XS_STRIDE;
    float*    T  = (float*)H1;   // reused after FFN2 reads complete (same-wave DS order)

    const int waveG = blockIdx.x * (blockDim.x >> 5) + wv;
    const int node  = waveG * 16 + r;

    float4 y[16];
    {
        const float* px = hVio + (size_t)node * DD + hi * 64;
        const float* pd = dh   + (size_t)node * DD + hi * 64;
        float s = 0.0f, s2 = 0.0f;
        #pragma unroll
        for (int i = 0; i < 16; ++i) {
            float4 a = *(const float4*)(px + i * 4);
            float4 d = *(const float4*)(pd + i * 4);
            float4 x;
            x.x = a.x + d.x; x.y = a.y + d.y; x.z = a.z + d.z; x.w = a.w + d.w;
            y[i] = x;
            s  += x.x + x.y + x.z + x.w;
            s2 += x.x * x.x + x.y * x.y + x.z * x.z + x.w * x.w;
        }
        s += __shfl_xor(s, 16); s2 += __shfl_xor(s2, 16);
        const float mean = s * (1.0f / 128.0f);
        const float inv  = rsqrtf(s2 * (1.0f / 128.0f) - mean * mean + EPSF);
        #pragma unroll
        for (int i = 0; i < 16; ++i) {
            float4 sv = *(const float4*)(ln1s + hi * 64 + i * 4);
            float4 bv = *(const float4*)(ln1b + hi * 64 + i * 4);
            y[i].x = sv.x * (y[i].x - mean) * inv + bv.x;
            y[i].y = sv.y * (y[i].y - mean) * inv + bv.y;
            y[i].z = sv.z * (y[i].z - mean) * inv + bv.z;
            y[i].w = sv.w * (y[i].w - mean) * inv + bv.w;
        }
    }
    // stage LN1 result as f16 A-tile
    {
        _Float16* xr = Xs + r * XS_STRIDE + hi * 64;
        #pragma unroll
        for (int i = 0; i < 8; ++i) {
            h8 h;
            h[0] = (_Float16)y[2*i].x;   h[1] = (_Float16)y[2*i].y;
            h[2] = (_Float16)y[2*i].z;   h[3] = (_Float16)y[2*i].w;
            h[4] = (_Float16)y[2*i+1].x; h[5] = (_Float16)y[2*i+1].y;
            h[6] = (_Float16)y[2*i+1].z; h[7] = (_Float16)y[2*i+1].w;
            *(h8*)(xr + i * 8) = h;
        }
    }
    // FFN1: (16x128) @ (128x512), B-frags streamed from global f16 (L2 resident)
    for (int g = 0; g < 4; ++g) {
        v8f C[8] = {};
        #pragma unroll
        for (int ks = 0; ks < 4; ++ks) {
            v16h A = frag_a(Xs, XS_STRIDE, r, ks * 32, hi);
            const _Float16* wb = WinH + (size_t)(ks * 32 + 8 * hi) * 512 + g * 128 + r;
            #pragma unroll
            for (int nt = 0; nt < 8; ++nt)
                C[nt] = wmma16(A, ld16(wb + nt * 16, 512), C[nt]);
        }
        #pragma unroll
        for (int nt = 0; nt < 8; ++nt) {
            const int c = g * 128 + nt * 16 + r;
            const float bias = b_in[c];
            #pragma unroll
            for (int v = 0; v < 8; ++v)
                H1[(v + 8 * hi) * H1_STRIDE + c] = (_Float16)gelu_exact(C[nt][v] + bias);
        }
    }
    // FFN2: (16x512) @ (512x128)
    v8f C2[8] = {};
    #pragma unroll
    for (int ks = 0; ks < 16; ++ks) {
        v16h A = frag_a(H1, H1_STRIDE, r, ks * 32, hi);
        const _Float16* wb = WoutH + (size_t)(ks * 32 + 8 * hi) * DD + r;
        #pragma unroll
        for (int nt = 0; nt < 8; ++nt)
            C2[nt] = wmma16(A, ld16(wb + nt * 16, DD), C2[nt]);
    }
    // transpose C layout -> row layout via LDS (f32)
    #pragma unroll
    for (int nt = 0; nt < 8; ++nt) {
        const int c = nt * 16 + r;
        const float bias = b_out[c];
        #pragma unroll
        for (int v = 0; v < 8; ++v)
            T[(v + 8 * hi) * T_STRIDE + c] = C2[nt][v] + bias;
    }
    // residual + LN2 + mask + store
    {
        float s = 0.0f, s2 = 0.0f;
        #pragma unroll
        for (int i = 0; i < 16; ++i) {
            float4 t = *(const float4*)(T + r * T_STRIDE + hi * 64 + i * 4);
            y[i].x += t.x; y[i].y += t.y; y[i].z += t.z; y[i].w += t.w;
            s  += y[i].x + y[i].y + y[i].z + y[i].w;
            s2 += y[i].x * y[i].x + y[i].y * y[i].y + y[i].z * y[i].z + y[i].w * y[i].w;
        }
        s += __shfl_xor(s, 16); s2 += __shfl_xor(s2, 16);
        const float mean = s * (1.0f / 128.0f);
        const float inv  = rsqrtf(s2 * (1.0f / 128.0f) - mean * mean + EPSF);
        const float mk = mask[node];
        float* po = hVio + (size_t)node * DD + hi * 64;
        #pragma unroll
        for (int i = 0; i < 16; ++i) {
            float4 sv = *(const float4*)(ln2s + hi * 64 + i * 4);
            float4 bv = *(const float4*)(ln2b + hi * 64 + i * 4);
            float4 o;
            o.x = (sv.x * (y[i].x - mean) * inv + bv.x) * mk;
            o.y = (sv.y * (y[i].y - mean) * inv + bv.y) * mk;
            o.z = (sv.z * (y[i].z - mean) * inv + bv.z) * mk;
            o.w = (sv.w * (y[i].w - mean) * inv + bv.w) * mk;
            *(float4*)(po + i * 4) = o;
        }
    }
}

// ---------------- output projection ----------------
__global__ void k_out(const float* __restrict__ hV, const float* __restrict__ outW,
                      const float* __restrict__ outb, float* __restrict__ out) {
    int node = blockIdx.x, v = threadIdx.x;
    if (v < VV) {
        float a = outb[v];
        const float* row = hV + (size_t)node * DD;
        #pragma unroll 8
        for (int c = 0; c < DD; ++c) a = fmaf(row[c], outW[c * VV + v], a);
        out[(size_t)node * VV + v] = a;
    }
}

// ---------------- host ----------------
extern "C" void kernel_launch(void* const* d_in, const int* in_sizes, int n_in,
                              void* d_out, int out_size, void* d_ws, size_t ws_size,
                              hipStream_t stream) {
    const int*   prev = (const int*)d_in[0];
    const float* hVin = (const float*)d_in[1];
    const float* hE   = (const float*)d_in[2];
    const int*   Eidx = (const int*)d_in[3];
    const float* mask = (const float*)d_in[4];
    const int*   dec  = (const int*)d_in[5];
    const float* embed= (const float*)d_in[6];
    const float* outW = (const float*)d_in[7];
    const float* outb = (const float*)d_in[8];
    const float* W1   = (const float*)d_in[9];
    const float* b1   = (const float*)d_in[10];
    const float* W2   = (const float*)d_in[11];
    const float* b2   = (const float*)d_in[12];
    const float* W3   = (const float*)d_in[13];
    const float* b3   = (const float*)d_in[14];
    const float* ln1s = (const float*)d_in[15];
    const float* ln1b = (const float*)d_in[16];
    const float* Win  = (const float*)d_in[17];
    const float* bin  = (const float*)d_in[18];
    const float* Wout = (const float*)d_in[19];
    const float* bout = (const float*)d_in[20];
    const float* ln2s = (const float*)d_in[21];
    const float* ln2b = (const float*)d_in[22];

    char* ws = (char*)d_ws;
    float*    hV    = (float*)(ws + 0);            // 2 MB
    float*    hV0   = (float*)(ws + 2097152);      // 2 MB
    float*    hS    = (float*)(ws + 4194304);      // 2 MB
    float*    dh    = (float*)(ws + 6291456);      // 2 MB
    _Float16* WinH  = (_Float16*)(ws + 8388608);   // 384 KB
    _Float16* WoutH = (_Float16*)(ws + 8781824);   // 384 KB
    int*      rank  = (int*)(ws + 9175040);        // 16 KB

    k_rank<<<(BB * NN + 255) / 256, 256, 0, stream>>>(dec, rank);
    k_prep<<<(BB * NN * DD + 255) / 256, 256, 0, stream>>>(prev, embed, hVin, hS, hV, hV0);
    k_cvtw<<<(LL * DD * 512 + 255) / 256, 256, 0, stream>>>(Win, Wout, WinH, WoutH);

    const size_t edgeLds = (size_t)(768 * EW_STRIDE) * 2 + 3 * DD * 4
                         + (size_t)4 * EDGE_PW;                    // 314,880 B <= 320 KB
    const size_t nodeLds = (size_t)4 * NODE_PW;                    // ~84 KB

    for (int l = 0; l < LL; ++l) {
        k_edge<<<512, 128, edgeLds, stream>>>(hE, Eidx, mask,
            W1 + (size_t)l * 512 * DD, b1 + l * DD,
            W2 + (size_t)l * DD * DD,  b2 + l * DD,
            W3 + (size_t)l * DD * DD,  b3 + l * DD,
            hS, hV, hV0, rank, dh);
        k_node<<<64, 128, nodeLds, stream>>>(hV, dh, mask,
            WinH  + (size_t)l * DD * 512, bin  + l * 512,
            WoutH + (size_t)l * 512 * DD, bout + l * DD,
            ln1s + l * DD, ln1b + l * DD, ln2s + l * DD, ln2b + l * DD);
    }
    k_out<<<BB * NN, 32, 0, stream>>>(hV, outW, outb, (float*)d_out);
}
